// SpatioTemporalAttention_81458349736718
// MI455X (gfx1250) — matrix operand order repaired and checked by reading, not compiled, not run on hardware
//
#include <hip/hip_runtime.h>
#include <math.h>

typedef __attribute__((ext_vector_type(16))) _Float16 v16h;
typedef __attribute__((ext_vector_type(8)))  _Float16 h8v;
typedef __attribute__((ext_vector_type(4)))  _Float16 h4v;
typedef __attribute__((ext_vector_type(8)))  float    v8f;

#define DIMC   512
#define NTOK   16384
#define NHEAD  8
#define DHEAD  64
#define FFIN   1365
#define FF2    2730
#define YNP    1376   // FFIN padded to multiple of 32 (f32 rows stay 16B aligned)

__device__ __forceinline__ float warp_sum(float v) {
#pragma unroll
  for (int off = 16; off > 0; off >>= 1) v += __shfl_xor(v, off, 32);
  return v;
}
__device__ __forceinline__ float warp_max(float v) {
#pragma unroll
  for (int off = 16; off > 0; off >>= 1) v = fmaxf(v, __shfl_xor(v, off, 32));
  return v;
}
__device__ __forceinline__ float siluf_(float x) { return x / (1.f + __expf(-x)); }
__device__ __forceinline__ float geluf_(float x) { return 0.5f * x * (1.f + erff(x * 0.70710678118654752f)); }

// ---------------------------------------------------------------------------
// Weight transpose + f16 convert: src[K][N] (f32) -> dst[N][Kp] (f16), zero pad.
// ---------------------------------------------------------------------------
__global__ void __launch_bounds__(256)
wtrans_f16(const float* __restrict__ src, _Float16* __restrict__ dst,
           int K, int N, int Kp)
{
  size_t o = (size_t)blockIdx.x * 256 + threadIdx.x;
  if (o >= (size_t)N * Kp) return;
  int k = (int)(o % Kp);
  int n = (int)(o / Kp);
  dst[o] = (k < K) ? (_Float16)src[(size_t)k * N + n] : (_Float16)0.f;
}

// ---------------------------------------------------------------------------
// Tiled GEMM: C[M,N] = A[M,K](f32, stride lda) @ B[K,N] given as BT[N][Kp](f16).
// Block = 128 threads (4 waves). Tile 64x64, K-step 32, double-buffered LDS.
// A staged via branchless float4->h4 (requires M % 64 == 0, lda >= Kp, A rows
// zero-padded in [K,lda)); B staged via global_load_async_to_lds_b128 (ASYNCcnt),
// BT zero-padded in [K,Kp); N-edge rows clamped (garbage cols masked at store).
// ---------------------------------------------------------------------------
__global__ void __launch_bounds__(128)
gemm_f16_wmma(const float* __restrict__ A, int lda,
              const _Float16* __restrict__ BT,
              float* __restrict__ C, int M, int N, int K, int Kp)
{
  __shared__ _Float16 As[2][64][40];   // [buf][row][k], 80B row stride
  __shared__ _Float16 Bs[2][64][40];   // [buf][col][k]
  const int tid  = threadIdx.x;
  const int lane = tid & 31;
  const int wv   = tid >> 5;
  const int m0   = blockIdx.y * 64;
  const int n0   = blockIdx.x * 64;

  const v8f zero8 = {0.f, 0.f, 0.f, 0.f, 0.f, 0.f, 0.f, 0.f};
  v8f acc[4];
#pragma unroll
  for (int nb = 0; nb < 4; ++nb) acc[nb] = zero8;

  const int rowA = (wv << 4) + (lane & 15);
  const int kbA  = (lane < 16) ? 0 : 8;    // A: K 0..7 & 16..23 | 8..15 & 24..31
  const int colB = lane & 15;
  const int kbB  = (lane < 16) ? 0 : 8;    // B: K 0..7 & 16..23 | 8..15 & 24..31

  auto stage = [&](int buf, int k0) {
    // A tile (64 rows x 32 k): branchless f32 -> f16
#pragma unroll
    for (int e = 0; e < 4; ++e) {
      int i4 = e * 128 + tid;          // 0..511 float4 chunks
      int r  = i4 >> 3, c4 = i4 & 7;
      float4 a = *(const float4*)&A[(size_t)(m0 + r) * lda + k0 + c4 * 4];
      h4v hv;
      hv[0] = (_Float16)a.x; hv[1] = (_Float16)a.y;
      hv[2] = (_Float16)a.z; hv[3] = (_Float16)a.w;
      *(h4v*)&As[buf][r][c4 * 4] = hv;
    }
    // B tile (64 cols x 32 k): async global->LDS, 16B per lane
#pragma unroll
    for (int e = 0; e < 2; ++e) {
      int ch = e * 128 + tid;          // 0..255 16B chunks
      int r  = ch >> 2, c8 = ch & 3;
      int gn = n0 + r;
      if (gn >= N) gn = N - 1;         // clamp; cols >= N masked at C store
      unsigned lds = (unsigned)(size_t)&Bs[buf][r][c8 * 8];
      unsigned long long ga =
          (unsigned long long)(size_t)&BT[(size_t)gn * Kp + k0 + c8 * 8];
      asm volatile("global_load_async_to_lds_b128 %0, %1, off"
                   :: "v"(lds), "v"(ga) : "memory");
    }
  };

  const int nIt = (K + 31) >> 5;
  stage(0, 0);
  asm volatile("s_wait_asynccnt 0" ::: "memory");
  __syncthreads();

  for (int it = 0; it < nIt; ++it) {
    const int cur = it & 1;
    // fragment loads for current tile (issued before next-tile staging so the
    // WMMA dscnt wait does not cover the staging stores)
    h8v alo = *(const h8v*)&As[cur][rowA][kbA];
    h8v ahi = *(const h8v*)&As[cur][rowA][kbA + 16];
    v16h af = __builtin_shufflevector(alo, ahi,
                0, 1, 2, 3, 4, 5, 6, 7, 8, 9, 10, 11, 12, 13, 14, 15);
    v16h bf[4];
#pragma unroll
    for (int nb = 0; nb < 4; ++nb) {
      const int col = (nb << 4) + colB;
      h8v blo = *(const h8v*)&Bs[cur][col][kbB];
      h8v bhi = *(const h8v*)&Bs[cur][col][kbB + 16];
      bf[nb] = __builtin_shufflevector(blo, bhi,
                 0, 1, 2, 3, 4, 5, 6, 7, 8, 9, 10, 11, 12, 13, 14, 15);
    }
    // overlap: stage next tile while this tile's WMMAs execute
    if (it + 1 < nIt) stage(cur ^ 1, (it + 1) << 5);
#pragma unroll
    for (int nb = 0; nb < 4; ++nb)
      acc[nb] = __builtin_amdgcn_wmma_f32_16x16x32_f16(false, af, false, bf[nb],
                                                       (short)0, acc[nb], false, false);
    if (it + 1 < nIt) {
      asm volatile("s_wait_asynccnt 0" ::: "memory");
      __syncthreads();   // publishes next tile; also drains this wave's DS ops
    }
  }

  // C layout: VGPR r -> row r (lanes 0-15) / r+8 (lanes 16-31), col = lane&15
  const int rbase = m0 + (wv << 4) + ((lane < 16) ? 0 : 8);
#pragma unroll
  for (int nb = 0; nb < 4; ++nb) {
    int gc = n0 + (nb << 4) + (lane & 15);
#pragma unroll
    for (int r = 0; r < 8; ++r) {
      int gr = rbase + r;
      if (gr < M && gc < N) C[(size_t)gr * N + gc] = acc[nb][r];
    }
  }
}

// ---------------------------------------------------------------------------
// LayerNorm over 512 channels, one wave per token.
// ---------------------------------------------------------------------------
__global__ void __launch_bounds__(256)
ln_kernel(const float* __restrict__ in, const float* __restrict__ w,
          const float* __restrict__ b, float* __restrict__ out)
{
  const int lane = threadIdx.x & 31;
  const int t = blockIdx.x * 8 + (threadIdx.x >> 5);
  const float* row = in + (size_t)t * DIMC + lane * 16;
  float v[16];
#pragma unroll
  for (int e = 0; e < 16; e += 4) {
    float4 a = *(const float4*)(row + e);
    v[e] = a.x; v[e + 1] = a.y; v[e + 2] = a.z; v[e + 3] = a.w;
  }
  float s = 0.f, ss = 0.f;
#pragma unroll
  for (int e = 0; e < 16; ++e) { s += v[e]; ss += v[e] * v[e]; }
  s = warp_sum(s); ss = warp_sum(ss);
  float mean = s * (1.f / 512.f);
  float var  = ss * (1.f / 512.f) - mean * mean;
  float inv  = rsqrtf(var + 1e-5f);
  float* orow = out + (size_t)t * DIMC;
#pragma unroll
  for (int e = 0; e < 16; ++e) {
    int c = lane * 16 + e;
    orow[c] = (v[e] - mean) * inv * w[c] + b[c];
  }
}

// LN + residual add (out may alias resid).
__global__ void __launch_bounds__(256)
ln_add_kernel(const float* __restrict__ in, const float* __restrict__ w,
              const float* __restrict__ b, const float* __restrict__ resid,
              float* __restrict__ out)
{
  const int lane = threadIdx.x & 31;
  const int t = blockIdx.x * 8 + (threadIdx.x >> 5);
  const float* row = in + (size_t)t * DIMC + lane * 16;
  float v[16];
#pragma unroll
  for (int e = 0; e < 16; e += 4) {
    float4 a = *(const float4*)(row + e);
    v[e] = a.x; v[e + 1] = a.y; v[e + 2] = a.z; v[e + 3] = a.w;
  }
  float s = 0.f, ss = 0.f;
#pragma unroll
  for (int e = 0; e < 16; ++e) { s += v[e]; ss += v[e] * v[e]; }
  s = warp_sum(s); ss = warp_sum(ss);
  float mean = s * (1.f / 512.f);
  float var  = ss * (1.f / 512.f) - mean * mean;
  float inv  = rsqrtf(var + 1e-5f);
  const float* rrow = resid + (size_t)t * DIMC;
  float* orow = out + (size_t)t * DIMC;
#pragma unroll
  for (int e = 0; e < 16; ++e) {
    int c = lane * 16 + e;
    orow[c] = (v[e] - mean) * inv * w[c] + b[c] + rrow[c];
  }
}

// ---------------------------------------------------------------------------
// Flash attention: one wave per (batch, head, query-row). MQA (shared K/V),
// null-KV at j==0, relative bias via precomputed CPB table.
// mode 0: spatial (N=1024, 32x32 grid), mode 1: temporal (N=16).
// ---------------------------------------------------------------------------
__global__ void __launch_bounds__(128)
attn_flash(const float* __restrict__ Q, const float* __restrict__ KV,
           const float* __restrict__ nullkv, const float* __restrict__ nullbias,
           const float* __restrict__ table, float* __restrict__ O,
           int N, int mode)
{
  __shared__ float sacc[4][32][DHEAD];
  __shared__ float se[4][32];
  const int lane = threadIdx.x & 31;
  const int wv   = threadIdx.x >> 5;
  const int row  = blockIdx.x * 4 + wv;   // (b*8 + h)*N + i
  const int i = row % N;
  const int h = (row / N) & 7;
  const int b = row / (N * 8);

  const float* qp = Q + ((size_t)(b * N + i) * DIMC + h * DHEAD);
  float q[DHEAD];
#pragma unroll
  for (int d = 0; d < DHEAD; d += 4) {
    float4 t = *(const float4*)(qp + d);
    q[d] = t.x * 0.125f; q[d + 1] = t.y * 0.125f;
    q[d + 2] = t.z * 0.125f; q[d + 3] = t.w * 0.125f;
  }
  float m = -3.0e38f, l = 0.f;
  float acc[DHEAD];
#pragma unroll
  for (int d = 0; d < DHEAD; ++d) acc[d] = 0.f;

  const int nk = N + 1;
  const float nbias = nullbias[h];
  for (int j = lane; j < nk; j += 32) {
    const float *kp, *vp;
    float bias;
    if (j == 0) { kp = nullkv; vp = nullkv + DHEAD; bias = nbias; }
    else {
      int jm = j - 1;
      const float* kv = KV + (size_t)(b * N + jm) * (2 * DHEAD);
      kp = kv; vp = kv + DHEAD;
      if (mode == 0) {
        int dy = (i >> 5) - (jm >> 5) + 31;
        int dx = (i & 31) - (jm & 31) + 31;
        bias = table[(dy * 63 + dx) * NHEAD + h];
      } else {
        bias = table[(i - jm + N - 1) * NHEAD + h];
      }
    }
    float s = bias;
#pragma unroll
    for (int d = 0; d < DHEAD; d += 4) {
      float4 kk = *(const float4*)(kp + d);
      s += q[d] * kk.x + q[d + 1] * kk.y + q[d + 2] * kk.z + q[d + 3] * kk.w;
    }
    float mn = fmaxf(m, s);
    float corr = __expf(m - mn);
    float p = __expf(s - mn);
    l = l * corr + p;
#pragma unroll
    for (int d = 0; d < DHEAD; d += 4) {
      float4 vv = *(const float4*)(vp + d);
      acc[d]     = acc[d]     * corr + p * vv.x;
      acc[d + 1] = acc[d + 1] * corr + p * vv.y;
      acc[d + 2] = acc[d + 2] * corr + p * vv.z;
      acc[d + 3] = acc[d + 3] * corr + p * vv.w;
    }
    m = mn;
  }

  float M = warp_max(m);
  float L = warp_sum(l * __expf(m - M));
  se[wv][lane] = __expf(m - M);
#pragma unroll
  for (int d = 0; d < DHEAD; ++d) sacc[wv][lane][d] = acc[d];
  __syncthreads();

  float o0 = 0.f, o1 = 0.f;
  for (int s = 0; s < 32; ++s) {
    float e = se[wv][s];
    o0 += sacc[wv][s][lane] * e;
    o1 += sacc[wv][s][lane + 32] * e;
  }
  float invL = 1.f / L;
  float* op = O + ((size_t)(b * N + i) * DIMC + h * DHEAD);
  op[lane]      = o0 * invL;
  op[lane + 32] = o1 * invL;
}

// ---------------------------------------------------------------------------
// CPB MLP bias table: one block per coordinate, 256 hidden units.
// in_dim==2: spatial (63x63 grid of (dy,dx)); in_dim==1: temporal (31 offsets).
// ---------------------------------------------------------------------------
__global__ void __launch_bounds__(256)
cpb_mlp_kernel(const float* __restrict__ w0, const float* __restrict__ b0,
               const float* __restrict__ w1, const float* __restrict__ b1,
               const float* __restrict__ w2, const float* __restrict__ b2,
               float* __restrict__ table, int in_dim)
{
  __shared__ float t0[256], t1[256];
  const int id = blockIdx.x;
  const int j  = threadIdx.x;
  float s;
  if (in_dim == 2) {
    float x0 = (float)(id / 63 - 31);
    float x1 = (float)(id % 63 - 31);
    s = x0 * w0[j] + x1 * w0[256 + j] + b0[j];
  } else {
    float x0 = (float)(id - 15);
    s = x0 * w0[j] + b0[j];
  }
  t0[j] = siluf_(s);
  __syncthreads();
  float s2 = b1[j];
  for (int k = 0; k < 256; ++k) s2 += t0[k] * w1[k * 256 + j];
  t1[j] = siluf_(s2);
  __syncthreads();
  if (j < NHEAD) {
    float s3 = b2[j];
    for (int k = 0; k < 256; ++k) s3 += t1[k] * w2[k * NHEAD + j];
    table[id * NHEAD + j] = s3;
  }
}

// ---------------------------------------------------------------------------
// GEGLU + time shift + channel norm; writes YN with zero-padded stride YNP.
// ---------------------------------------------------------------------------
__global__ void __launch_bounds__(256)
glu_shift_norm(const float* __restrict__ T1, const float* __restrict__ g,
               float* __restrict__ YN)
{
  __shared__ float yls[FFIN];
  __shared__ float rs[8], rss[8];
  const int t = blockIdx.x;
  const int f = t & 15;
  for (int c = threadIdx.x; c < FFIN; c += 256) {
    int src = (c < 683) ? t : (f > 0 ? t - 1 : -1);   // y2 channels come from f-1
    float y = 0.f;
    if (src >= 0) {
      const float* rowp = T1 + (size_t)src * FF2;
      float a  = rowp[c];
      float gt = rowp[FFIN + c];
      y = a * geluf_(gt);
    }
    yls[c] = y;
  }
  __syncthreads();
  float s = 0.f, ss = 0.f;
  for (int c = threadIdx.x; c < FFIN; c += 256) { float y = yls[c]; s += y; ss += y * y; }
  s = warp_sum(s); ss = warp_sum(ss);
  const int lane = threadIdx.x & 31, wv = threadIdx.x >> 5;
  if (lane == 0) { rs[wv] = s; rss[wv] = ss; }
  __syncthreads();
  float S = 0.f, SS = 0.f;
#pragma unroll
  for (int w = 0; w < 8; ++w) { S += rs[w]; SS += rss[w]; }
  float mean = S * (1.f / FFIN);
  float var  = fmaxf(SS * (1.f / FFIN) - mean * mean, 1e-5f);
  float inv  = rsqrtf(var);
  float* orow = YN + (size_t)t * YNP;
  for (int c = threadIdx.x; c < FFIN; c += 256) orow[c] = (yls[c] - mean) * inv * g[c];
  for (int c = FFIN + threadIdx.x; c < YNP; c += 256) orow[c] = 0.f;   // pad
}

// ---------------------------------------------------------------------------
// Layout permutes.
// ---------------------------------------------------------------------------
// x (1,512,16,32,32) -> xs[t=f*1024+hw][c]
__global__ void __launch_bounds__(256)
permute_in(const float* __restrict__ x, float* __restrict__ xs)
{
  size_t o = (size_t)blockIdx.x * 256 + threadIdx.x;
  int c = (int)(o & 511);
  size_t t = o >> 9;
  xs[o] = x[(size_t)c * NTOK + t];
}
// xs[f*1024+hw][c] -> xt[hw*16+f][c]
__global__ void __launch_bounds__(256)
permute_s_to_t(const float* __restrict__ xs, float* __restrict__ xt)
{
  size_t o = (size_t)blockIdx.x * 256 + threadIdx.x;
  int c = (int)(o & 511);
  size_t tt = o >> 9;
  int f  = (int)(tt & 15);
  int hw = (int)(tt >> 4);
  xt[o] = xs[((size_t)(f * 1024 + hw) << 9) + c];
}
// out[c][f*1024+hw] = FO[hw*16+f][c] + xt[hw*16+f][c]
__global__ void __launch_bounds__(256)
final_out(const float* __restrict__ FO, const float* __restrict__ xt,
          float* __restrict__ out)
{
  size_t o = (size_t)blockIdx.x * 256 + threadIdx.x;
  size_t t = o & (NTOK - 1);
  int c  = (int)(o >> 14);
  int f  = (int)(t >> 10);
  int hw = (int)(t & 1023);
  size_t idx = ((size_t)(hw * 16 + f) << 9) + c;
  out[o] = FO[idx] + xt[idx];
}

// ---------------------------------------------------------------------------
extern "C" void kernel_launch(void* const* d_in, const int* in_sizes, int n_in,
                              void* d_out, int out_size, void* d_ws, size_t ws_size,
                              hipStream_t stream)
{
  (void)in_sizes; (void)n_in; (void)out_size; (void)ws_size;
  const float* x        = (const float*)d_in[0];
  const float* sa_lnw   = (const float*)d_in[1];
  const float* sa_lnb   = (const float*)d_in[2];
  const float* sa_wq    = (const float*)d_in[3];
  const float* sa_wkv   = (const float*)d_in[4];
  const float* sa_nkv   = (const float*)d_in[5];
  const float* sa_nb    = (const float*)d_in[6];
  const float* sa_wout  = (const float*)d_in[7];
  const float* sa_olnw  = (const float*)d_in[8];
  const float* sa_olnb  = (const float*)d_in[9];
  const float* ta_lnw   = (const float*)d_in[10];
  const float* ta_lnb   = (const float*)d_in[11];
  const float* ta_wq    = (const float*)d_in[12];
  const float* ta_wkv   = (const float*)d_in[13];
  const float* ta_nkv   = (const float*)d_in[14];
  const float* ta_nb    = (const float*)d_in[15];
  const float* ta_wout  = (const float*)d_in[16];
  const float* ta_olnw  = (const float*)d_in[17];
  const float* ta_olnb  = (const float*)d_in[18];
  const float* sp_w0 = (const float*)d_in[19];
  const float* sp_b0 = (const float*)d_in[20];
  const float* sp_w1 = (const float*)d_in[21];
  const float* sp_b1 = (const float*)d_in[22];
  const float* sp_w2 = (const float*)d_in[23];
  const float* sp_b2 = (const float*)d_in[24];
  const float* tp_w0 = (const float*)d_in[25];
  const float* tp_b0 = (const float*)d_in[26];
  const float* tp_w1 = (const float*)d_in[27];
  const float* tp_b1 = (const float*)d_in[28];
  const float* tp_w2 = (const float*)d_in[29];
  const float* tp_b2 = (const float*)d_in[30];
  const float* ff_win  = (const float*)d_in[31];
  const float* ff_g    = (const float*)d_in[32];
  const float* ff_wout = (const float*)d_in[33];
  float* out = (float*)d_out;

  // workspace layout (bytes)
  const size_t SZ_TC = (size_t)NTOK * DIMC * 4;       // 32 MB
  const size_t SZ_KV = (size_t)NTOK * 128 * 4;        // 8 MB
  const size_t SZ_T1 = (size_t)NTOK * FF2 * 4;        // ~171 MB
  char* ws = (char*)d_ws;
  float* xs  = (float*)(ws);                          // spatial tokens / xs'
  float* xt  = (float*)(ws + SZ_TC);                  // temporal tokens / xt'
  float* lnb = (float*)(ws + 2 * SZ_TC);              // LN buffer, later FO
  float* Qb  = (float*)(ws + 3 * SZ_TC);
  float* KVb = (float*)(ws + 4 * SZ_TC);
  float* Ob  = (float*)(ws + 4 * SZ_TC + SZ_KV);
  float* Pb  = (float*)(ws + 5 * SZ_TC + SZ_KV);
  float* T1b = (float*)(ws + 6 * SZ_TC + SZ_KV);
  float* spT = (float*)(ws + 6 * SZ_TC + SZ_KV + SZ_T1);
  float* tpT = (float*)((char*)spT + (size_t)3969 * 8 * 4);
  // transposed f16 weights (BT[N][Kp])
  char* wb = (char*)tpT + 992;
  _Float16* wq_sT  = (_Float16*)(wb);                          wb += (size_t)512 * 512 * 2;
  _Float16* wkv_sT = (_Float16*)(wb);                          wb += (size_t)128 * 512 * 2;
  _Float16* wo_sT  = (_Float16*)(wb);                          wb += (size_t)512 * 512 * 2;
  _Float16* wq_tT  = (_Float16*)(wb);                          wb += (size_t)512 * 512 * 2;
  _Float16* wkv_tT = (_Float16*)(wb);                          wb += (size_t)128 * 512 * 2;
  _Float16* wo_tT  = (_Float16*)(wb);                          wb += (size_t)512 * 512 * 2;
  _Float16* winT   = (_Float16*)(wb);                          wb += (size_t)FF2 * 512 * 2;
  _Float16* woutT  = (_Float16*)(wb);                          wb += (size_t)512 * YNP * 2;
  float* YNb = Qb;        // reuse Q/KV/O region in FF phase (needs 90 MB < 104 MB)
  float* FOb = lnb;       // reuse LN buffer for final GEMM output

  auto wtrans = [&](const float* W, _Float16* WT, int K, int N) {
    int Kp = (K + 31) & ~31;
    size_t tot = (size_t)N * Kp;
    wtrans_f16<<<(unsigned)((tot + 255) / 256), 256, 0, stream>>>(W, WT, K, N, Kp);
  };
  auto gemm = [&](const float* A, int lda, const _Float16* BT, float* Cm,
                  int M, int N, int K) {
    int Kp = (K + 31) & ~31;
    dim3 g((N + 63) / 64, (M + 63) / 64);
    gemm_f16_wmma<<<g, 128, 0, stream>>>(A, lda, BT, Cm, M, N, K, Kp);
  };

  // weight pre-transpose (f32 -> f16, [K][N] -> [N][Kp])
  wtrans(sa_wq,   wq_sT,  512, 512);
  wtrans(sa_wkv,  wkv_sT, 512, 128);
  wtrans(sa_wout, wo_sT,  512, 512);
  wtrans(ta_wq,   wq_tT,  512, 512);
  wtrans(ta_wkv,  wkv_tT, 512, 128);
  wtrans(ta_wout, wo_tT,  512, 512);
  wtrans(ff_win,  winT,   512, FF2);
  wtrans(ff_wout, woutT,  FFIN, 512);

  // bias tables
  cpb_mlp_kernel<<<3969, 256, 0, stream>>>(sp_w0, sp_b0, sp_w1, sp_b1, sp_w2, sp_b2, spT, 2);
  cpb_mlp_kernel<<<31,   256, 0, stream>>>(tp_w0, tp_b0, tp_w1, tp_b1, tp_w2, tp_b2, tpT, 1);

  // ---- spatial attention ----
  permute_in<<<NTOK * DIMC / 256, 256, 0, stream>>>(x, xs);
  ln_kernel<<<NTOK / 8, 256, 0, stream>>>(xs, sa_lnw, sa_lnb, lnb);
  gemm(lnb, 512, wq_sT,  Qb,  NTOK, 512, 512);
  gemm(lnb, 512, wkv_sT, KVb, NTOK, 128, 512);
  attn_flash<<<(16 * NHEAD * 1024) / 4, 128, 0, stream>>>(Qb, KVb, sa_nkv, sa_nb, spT, Ob, 1024, 0);
  gemm(Ob, 512, wo_sT, Pb, NTOK, 512, 512);
  ln_add_kernel<<<NTOK / 8, 256, 0, stream>>>(Pb, sa_olnw, sa_olnb, xs, xs);

  // ---- temporal attention ----
  permute_s_to_t<<<NTOK * DIMC / 256, 256, 0, stream>>>(xs, xt);
  ln_kernel<<<NTOK / 8, 256, 0, stream>>>(xt, ta_lnw, ta_lnb, lnb);
  gemm(lnb, 512, wq_tT,  Qb,  NTOK, 512, 512);
  gemm(lnb, 512, wkv_tT, KVb, NTOK, 128, 512);
  attn_flash<<<(1024 * NHEAD * 16) / 4, 128, 0, stream>>>(Qb, KVb, ta_nkv, ta_nb, tpT, Ob, 16, 1);
  gemm(Ob, 512, wo_tT, Pb, NTOK, 512, 512);
  ln_add_kernel<<<NTOK / 8, 256, 0, stream>>>(Pb, ta_olnw, ta_olnb, xt, xt);

  // ---- feed-forward (GEGLU + time shift + channel norm) ----
  gemm(xt, 512, winT, T1b, NTOK, FF2, 512);
  glu_shift_norm<<<NTOK, 256, 0, stream>>>(T1b, ff_g, YNb);
  gemm(YNb, YNP, woutT, FOb, NTOK, 512, FFIN);
  final_out<<<NTOK * DIMC / 256, 256, 0, stream>>>(FOb, xt, out);
}
